// HAN_20985210209011
// MI455X (gfx1250) — matrix operand order repaired and checked by reading, not compiled, not run on hardware
//
#include <hip/hip_runtime.h>
#include <math.h>

typedef __bf16 bf16_t;
typedef __attribute__((ext_vector_type(16))) __bf16 bf16x16;
typedef __attribute__((ext_vector_type(8)))  __bf16 bf16x8;
typedef __attribute__((ext_vector_type(8)))  float  f32x8;

// ---------------------------------------------------------------------------
// WMMA tile loaders (wave32, V_WMMA_F32_16X16X32_BF16; contiguous k order, shared by A and B)
// A: 16x32 (MxK). lane<16: m=lane, elems = A[m][k0+0..15]
//                 lane>=16: m=lane-16, elems = A[m][k0+16..31]
// ---------------------------------------------------------------------------
__device__ inline bf16x16 load_a_tile(const bf16_t* base, int kstride, int k0, int lane) {
    const int half = lane >> 4;
    const int m    = lane & 15;
    const bf16_t* p = base + m * kstride + k0 + 16 * half;
    bf16x8 lo = *(const bf16x8*)p;
    bf16x8 hi = *(const bf16x8*)(p + 8);
    bf16x16 r;
#pragma unroll
    for (int i = 0; i < 8; ++i) { r[i] = lo[i]; r[i + 8] = hi[i]; }
    return r;
}

// B: 32x16 (KxN), sourced from row-major weight W[n][k] (i.e. B = W^T).
// lane: n = n0 + lane%16; elems = W[n][k0 + 16*(lane/16) + 0..15]  (32B contiguous)
__device__ inline bf16x16 load_b_tile(const bf16_t* W, int kstride, int n0, int k0, int lane) {
    const int half = lane >> 4;
    const int n    = n0 + (lane & 15);
    return *(const bf16x16*)(W + (size_t)n * kstride + k0 + 16 * half);
}

// ---------------------------------------------------------------------------
// f32 -> bf16 weight conversion
// ---------------------------------------------------------------------------
__global__ void han_cvt_bf16(const float* __restrict__ src, bf16_t* __restrict__ dst, int n) {
    int i = blockIdx.x * blockDim.x + threadIdx.x;
    int stride = gridDim.x * blockDim.x;
    for (; i < n; i += stride) dst[i] = (bf16_t)src[i];
}

// ---------------------------------------------------------------------------
// Input projection: out[M,3072] = gather(Asrc)[M,K] @ W^T + bias
// One block = 16 rows (staged to LDS as bf16); 8 waves x 24 N-tiles of 16.
// tokens==nullptr -> identity row gather. K is 1024 or 2048.
// ---------------------------------------------------------------------------
__global__ void __launch_bounds__(256) han_proj(
    float* __restrict__ out, const float* __restrict__ Asrc,
    const int* __restrict__ tokens, const bf16_t* __restrict__ Wbf,
    const float* __restrict__ bias, int K)
{
    __shared__ bf16_t Asta[16 * 2048];   // 64KB max
    const int tid = threadIdx.x;
    const int lane = tid & 31;
    const int wave = tid >> 5;
    const int half = lane >> 4;
    const int rowbase = blockIdx.x * 16;

    const int total = 16 * K;
    for (int idx = tid; idx < total; idx += 256) {
        int row = idx / K;
        int col = idx - row * K;
        int rg = rowbase + row;
        int srow = tokens ? tokens[rg] : rg;
        Asta[idx] = (bf16_t)Asrc[(size_t)srow * K + col];
    }
    __syncthreads();

    for (int q = 0; q < 24; ++q) {
        const int n0 = (wave * 24 + q) * 16;
        f32x8 acc = {};
        for (int k0 = 0; k0 < K; k0 += 32) {
            bf16x16 a = load_a_tile(Asta, K, k0, lane);
            bf16x16 b = load_b_tile(Wbf, K, n0, k0, lane);
            acc = __builtin_amdgcn_wmma_f32_16x16x32_bf16(false, a, false, b,
                                                          (short)0, acc, false, false);
        }
        const int n = n0 + (lane & 15);
        const float bs = bias[n];
#pragma unroll
        for (int v = 0; v < 8; ++v) {
            const int m = v + 8 * half;
            out[(size_t)(rowbase + m) * 3072 + n] = acc[v] + bs;
        }
    }
}

// ---------------------------------------------------------------------------
// Word-level GRU recurrence. 12 blocks x 512 threads (16 waves).
// Block b owns 16 sequences seq = b*16 + m; seq = dir*96 + s.
// Hidden state: f32 in registers (owning wave), bf16 double-buffered in LDS
// for the cross-wave WMMA A operand. One barrier per time step.
// Each wave owns 4 j-tiles; per tile: 3 gate accumulators (r,z,n).
// ---------------------------------------------------------------------------
__global__ void __launch_bounds__(512) han_wordgru(
    const float* __restrict__ xW,    // [96*96, 3072], row = s*96 + t
    const bf16_t* __restrict__ Whh,  // [3072, 1024] bf16 row-major
    const float* __restrict__ bhh,   // [3072]
    float* __restrict__ f_hs,        // [96, 96, 1024]
    float* __restrict__ b_hs)        // [96, 96, 1024]
{
    __shared__ bf16_t Hbf[2][16 * 1024];  // 64KB
    const int tid = threadIdx.x, lane = tid & 31, wave = tid >> 5;
    const int half = lane >> 4;

    for (int i = tid; i < 2 * 16 * 1024; i += 512) Hbf[0][i] = (bf16_t)0.0f;
    __syncthreads();

    const int seq_base = blockIdx.x * 16;
    const int dir = (seq_base >= 96) ? 1 : 0;
    const int s_base = seq_base - 96 * dir;
    float* hs_out = dir ? b_hs : f_hs;

    float hreg[4][8];
#pragma unroll
    for (int q = 0; q < 4; ++q)
#pragma unroll
        for (int v = 0; v < 8; ++v) hreg[q][v] = 0.0f;

    for (int t = 0; t < 96; ++t) {
        const int tt = dir ? (95 - t) : t;
        const int rb = t & 1;
        const int wb = rb ^ 1;

        for (int q = 0; q < 4; ++q) {
            const int j0 = (wave * 4 + q) * 16;
            f32x8 ar = {}, az = {}, an = {};
            for (int k0 = 0; k0 < 1024; k0 += 32) {
                bf16x16 a  = load_a_tile(Hbf[rb], 1024, k0, lane);
                bf16x16 br = load_b_tile(Whh, 1024, j0,        k0, lane);
                ar = __builtin_amdgcn_wmma_f32_16x16x32_bf16(false, a, false, br, (short)0, ar, false, false);
                bf16x16 bz = load_b_tile(Whh, 1024, j0 + 1024, k0, lane);
                az = __builtin_amdgcn_wmma_f32_16x16x32_bf16(false, a, false, bz, (short)0, az, false, false);
                bf16x16 bn = load_b_tile(Whh, 1024, j0 + 2048, k0, lane);
                an = __builtin_amdgcn_wmma_f32_16x16x32_bf16(false, a, false, bn, (short)0, an, false, false);
            }
            const int n = j0 + (lane & 15);
            const float br_ = bhh[n], bz_ = bhh[n + 1024], bn_ = bhh[n + 2048];
#pragma unroll
            for (int v = 0; v < 8; ++v) {
                const int m = v + 8 * half;
                const int s = s_base + m;
                const float* xrow = xW + ((size_t)s * 96 + tt) * 3072;
                const float r  = 1.0f / (1.0f + expf(-(xrow[n]        + ar[v] + br_)));
                const float z  = 1.0f / (1.0f + expf(-(xrow[n + 1024] + az[v] + bz_)));
                const float ng = tanhf(xrow[n + 2048] + r * (an[v] + bn_));
                const float hn = (1.0f - z) * ng + z * hreg[q][v];
                hreg[q][v] = hn;
                Hbf[wb][m * 1024 + n] = (bf16_t)hn;
                hs_out[((size_t)s * 96 + tt) * 1024 + n] = hn;
            }
        }
        __syncthreads();  // all reads of Hbf[rb] done before next step overwrites it
    }
}

// ---------------------------------------------------------------------------
// Word attention: unnormalized exp scores, weighted sum -> sent_vecs[96,2048]
// ---------------------------------------------------------------------------
__global__ void __launch_bounds__(128) han_wattn(
    const float* __restrict__ f_hs, const float* __restrict__ b_hs,
    const float* __restrict__ w, const float* __restrict__ b,
    float* __restrict__ sent)
{
    __shared__ float sc[96];
    const int s = blockIdx.x, tid = threadIdx.x;
    if (tid < 96) {
        const float* f  = f_hs + ((size_t)s * 96 + tid) * 1024;
        const float* bk = b_hs + ((size_t)s * 96 + tid) * 1024;
        float acc = b[0];
        for (int j = 0; j < 1024; ++j) acc += f[j] * w[j] + bk[j] * w[1024 + j];
        sc[tid] = expf(acc);
    }
    __syncthreads();
    for (int d = tid; d < 2048; d += 128) {
        const float* src = (d < 1024) ? f_hs : b_hs;
        const int j = d & 1023;
        float acc = 0.0f;
        for (int t = 0; t < 96; ++t)
            acc += sc[t] * src[((size_t)s * 96 + t) * 1024 + j];
        sent[(size_t)s * 2048 + d] = acc;
    }
}

// ---------------------------------------------------------------------------
// Sentence GRU: one block, M=16 tile with row0 = forward, row1 = backward,
// rows 2..15 computed but ignored. Same WMMA recurrence as the word GRU.
// ---------------------------------------------------------------------------
__global__ void __launch_bounds__(512) han_sentgru(
    const float* __restrict__ sxW,   // [96, 3072]
    const bf16_t* __restrict__ Whh,  // [3072, 1024]
    const float* __restrict__ bhh,
    float* __restrict__ sf, float* __restrict__ sb)  // [96,1024] each
{
    __shared__ bf16_t Hbf[2][16 * 1024];
    const int tid = threadIdx.x, lane = tid & 31, wave = tid >> 5;
    const int half = lane >> 4;

    for (int i = tid; i < 2 * 16 * 1024; i += 512) Hbf[0][i] = (bf16_t)0.0f;
    __syncthreads();

    float hreg[4][8];
#pragma unroll
    for (int q = 0; q < 4; ++q)
#pragma unroll
        for (int v = 0; v < 8; ++v) hreg[q][v] = 0.0f;

    for (int t = 0; t < 96; ++t) {
        const int rb = t & 1;
        const int wb = rb ^ 1;
        for (int q = 0; q < 4; ++q) {
            const int j0 = (wave * 4 + q) * 16;
            f32x8 ar = {}, az = {}, an = {};
            for (int k0 = 0; k0 < 1024; k0 += 32) {
                bf16x16 a  = load_a_tile(Hbf[rb], 1024, k0, lane);
                bf16x16 br = load_b_tile(Whh, 1024, j0,        k0, lane);
                ar = __builtin_amdgcn_wmma_f32_16x16x32_bf16(false, a, false, br, (short)0, ar, false, false);
                bf16x16 bz = load_b_tile(Whh, 1024, j0 + 1024, k0, lane);
                az = __builtin_amdgcn_wmma_f32_16x16x32_bf16(false, a, false, bz, (short)0, az, false, false);
                bf16x16 bn = load_b_tile(Whh, 1024, j0 + 2048, k0, lane);
                an = __builtin_amdgcn_wmma_f32_16x16x32_bf16(false, a, false, bn, (short)0, an, false, false);
            }
            const int n = j0 + (lane & 15);
            const float br_ = bhh[n], bz_ = bhh[n + 1024], bn_ = bhh[n + 2048];
#pragma unroll
            for (int v = 0; v < 8; ++v) {
                const int m = v + 8 * half;
                const int tt = (m == 1) ? (95 - t) : t;   // row1 walks backward
                const float* xrow = sxW + (size_t)tt * 3072;
                const float r  = 1.0f / (1.0f + expf(-(xrow[n]        + ar[v] + br_)));
                const float z  = 1.0f / (1.0f + expf(-(xrow[n + 1024] + az[v] + bz_)));
                const float ng = tanhf(xrow[n + 2048] + r * (an[v] + bn_));
                const float hn = (1.0f - z) * ng + z * hreg[q][v];
                hreg[q][v] = hn;
                Hbf[wb][m * 1024 + n] = (bf16_t)hn;
                if (m == 0)      sf[(size_t)t * 1024 + n] = hn;
                else if (m == 1) sb[(size_t)(95 - t) * 1024 + n] = hn;
            }
        }
        __syncthreads();
    }
}

// ---------------------------------------------------------------------------
// Sentence attention -> document vector [2048]
// ---------------------------------------------------------------------------
__global__ void __launch_bounds__(128) han_sattn(
    const float* __restrict__ sf, const float* __restrict__ sb,
    const float* __restrict__ w, const float* __restrict__ b,
    float* __restrict__ out)
{
    __shared__ float sc[96];
    const int tid = threadIdx.x;
    if (tid < 96) {
        const float* f  = sf + (size_t)tid * 1024;
        const float* bk = sb + (size_t)tid * 1024;
        float acc = b[0];
        for (int j = 0; j < 1024; ++j) acc += f[j] * w[j] + bk[j] * w[1024 + j];
        sc[tid] = expf(acc);
    }
    __syncthreads();
    for (int d = tid; d < 2048; d += 128) {
        const float* src = (d < 1024) ? sf : sb;
        const int j = d & 1023;
        float acc = 0.0f;
        for (int s = 0; s < 96; ++s) acc += sc[s] * src[(size_t)s * 1024 + j];
        out[d] = acc;
    }
}

// ---------------------------------------------------------------------------
// Host launcher
// ---------------------------------------------------------------------------
extern "C" void kernel_launch(void* const* d_in, const int* in_sizes, int n_in,
                              void* d_out, int out_size, void* d_ws, size_t ws_size,
                              hipStream_t stream)
{
    (void)in_sizes; (void)n_in; (void)out_size; (void)ws_size;
    const int*   tokens = (const int*)  d_in[0];
    const float* emb    = (const float*)d_in[1];
    const float* wWih   = (const float*)d_in[2];
    const float* wWhh   = (const float*)d_in[3];
    const float* wbih   = (const float*)d_in[4];
    const float* wbhh   = (const float*)d_in[5];
    const float* sWih   = (const float*)d_in[6];
    const float* sWhh   = (const float*)d_in[7];
    const float* sbih   = (const float*)d_in[8];
    const float* sbhh   = (const float*)d_in[9];
    const float* wctxw  = (const float*)d_in[10];
    const float* wctxb  = (const float*)d_in[11];
    const float* sctxw  = (const float*)d_in[12];
    const float* sctxb  = (const float*)d_in[13];
    float* out = (float*)d_out;

    char* p = (char*)d_ws;
    auto alloc = [&](size_t bytes) -> char* {
        char* r = p;
        p += (bytes + 255) & ~(size_t)255;
        return r;
    };
    float*  xW    = (float*) alloc((size_t)9216 * 3072 * 4);
    float*  f_hs  = (float*) alloc((size_t)96 * 96 * 1024 * 4);
    float*  b_hs  = (float*) alloc((size_t)96 * 96 * 1024 * 4);
    float*  sent  = (float*) alloc((size_t)96 * 2048 * 4);
    float*  sxW   = (float*) alloc((size_t)96 * 3072 * 4);
    float*  sf    = (float*) alloc((size_t)96 * 1024 * 4);
    float*  sb    = (float*) alloc((size_t)96 * 1024 * 4);
    bf16_t* WihB  = (bf16_t*)alloc((size_t)3072 * 1024 * 2);
    bf16_t* WhhB  = (bf16_t*)alloc((size_t)3072 * 1024 * 2);
    bf16_t* sWihB = (bf16_t*)alloc((size_t)3072 * 2048 * 2);
    bf16_t* sWhhB = (bf16_t*)alloc((size_t)3072 * 1024 * 2);

    han_cvt_bf16<<<512, 256, 0, stream>>>(wWih, WihB,  3072 * 1024);
    han_cvt_bf16<<<512, 256, 0, stream>>>(wWhh, WhhB,  3072 * 1024);
    han_cvt_bf16<<<512, 256, 0, stream>>>(sWih, sWihB, 3072 * 2048);
    han_cvt_bf16<<<512, 256, 0, stream>>>(sWhh, sWhhB, 3072 * 1024);

    // xW[s*96+t] = emb[tokens[s,t]] @ Wih^T + bih
    han_proj<<<576, 256, 0, stream>>>(xW, emb, tokens, WihB, wbih, 1024);
    // word-level bi-GRU (192 sequences, 12 blocks of 16)
    han_wordgru<<<12, 512, 0, stream>>>(xW, WhhB, wbhh, f_hs, b_hs);
    // word attention -> sent_vecs [96, 2048]
    han_wattn<<<96, 128, 0, stream>>>(f_hs, b_hs, wctxw, wctxb, sent);
    // sentence input projection: sxW = sent @ sWih^T + sbih
    han_proj<<<6, 256, 0, stream>>>(sxW, sent, nullptr, sWihB, sbih, 2048);
    // sentence bi-GRU
    han_sentgru<<<1, 512, 0, stream>>>(sxW, sWhhB, sbhh, sf, sb);
    // sentence attention -> document vector
    han_sattn<<<1, 128, 0, stream>>>(sf, sb, sctxw, sctxb, out);
}
